// Encoder_88691074663016
// MI455X (gfx1250) — matrix operand order repaired and checked by reading, not compiled, not run on hardware
//
#include <hip/hip_runtime.h>
#include <hip/hip_bf16.h>

// ---------------- problem constants ----------------
#define B_   128
#define T_   60
#define V_   60000
#define E_   300
#define H_   300
#define C_   50000
#define S_   2186
#define L_   45
#define R_   1000
#define D2_  600
#define G4H_ 1200

#define EP_   320   // E padded to 32
#define D2P_  608   // D2 padded to 32
#define SP_   2208  // S padded
#define RP_   1024  // R padded
#define LP_   64    // L padded

#define NEG_POS_ (-10000000.0f)
#define NEG_MSK_ (-1000000.0f)

typedef __attribute__((ext_vector_type(8)))  float        v8f;
typedef __attribute__((ext_vector_type(16))) __bf16       v16bf;
typedef __attribute__((ext_vector_type(4)))  unsigned int u32x4;

union Vb {
    v16bf v;
    unsigned short u[16];
    u32x4 q[2];
};
union Pk {
    unsigned short s[8];
    u32x4 q;
};

__device__ __forceinline__ unsigned short f2bf(float f) {
    unsigned int u = __float_as_uint(f);
    unsigned int r = (u + 0x7FFFu + ((u >> 16) & 1u)) >> 16;
    return (unsigned short)r;
}
__device__ __forceinline__ float sigf(float x) { return 1.0f / (1.0f + expf(-x)); }

// ---------------- utility zero kernels ----------------
__global__ void zero_u16_kernel(unsigned short* p, size_t n) {
    size_t i = (size_t)blockIdx.x * blockDim.x + threadIdx.x;
    size_t st = (size_t)gridDim.x * blockDim.x;
    for (; i < n; i += st) p[i] = 0;
}
__global__ void zero_f32_kernel(float* p, size_t n) {
    size_t i = (size_t)blockIdx.x * blockDim.x + threadIdx.x;
    size_t st = (size_t)gridDim.x * blockDim.x;
    for (; i < n; i += st) p[i] = 0.0f;
}

// ---------------- embedding gather + mask ----------------
__global__ __launch_bounds__(EP_) void embed_kernel(const int* __restrict__ x,
                                                    const float* __restrict__ emb,
                                                    unsigned short* __restrict__ xemb_bf,
                                                    float* __restrict__ mask) {
    int bt = blockIdx.x;           // 0..B*T-1
    int e  = threadIdx.x;          // 0..319
    int xv = x[bt];
    if (e == 0) mask[bt] = (xv > 0) ? 1.0f : 0.0f;
    unsigned short v = 0;
    if (e < E_) v = f2bf(emb[(size_t)xv * E_ + e]);
    xemb_bf[(size_t)bt * EP_ + e] = v;
}

// ---------------- generic bf16 WMMA GEMM ----------------
// C[M,N] = act( A[M,K(bf16,padded lda)] * W[N,K(fp32)]^T + bias[N] )
// Block: 256 threads = 8 waves; block tile 64(M) x 64(N); wave tile 16x32.
//
// LDS layouts: As2 is fragment-native so A fragment reads are 2x ds_load_b128:
//  As2: per M-row, K-groups of 8 permuted [g0,g2,g1,g3] -> lane reads 32
//       contiguous bytes at row*64B + (lane>=16)*32B.
//  Bs2: k-major [k][n] (64 n per k row) -> lane (=n) reads 16 strided
//       halves at k*128B + n*2B (k in table order of its lane half).
// W staging: lane = k, thread owns 8 consecutive n -> each unrolled iteration
// is a coalesced 128B wave read of one W row; the 8 bf16 results are
// contiguous in Bs2 -> single ds_store_b128 per thread. One uniform
// fast/slow branch per tile (no per-element exec churn).
__global__ __launch_bounds__(256) void gemm_bf16_kernel(
    const unsigned short* __restrict__ A, int lda,
    const float* __restrict__ W, int ldw,
    const float* __restrict__ bias,
    float* __restrict__ Cout,
    int M, int N, int K, int act) {
    __shared__ alignas(16) unsigned short As2[64 * 32];
    __shared__ alignas(16) unsigned short Bs2[32 * 64];
    const int tid  = threadIdx.x;
    const int lane = tid & 31;
    const int w    = tid >> 5;
    const int mBase = blockIdx.y * 64;
    const int nBase = blockIdx.x * 64;
    const int moff  = (w & 3) * 16;
    const int noff  = (w >> 2) * 32;

    // A staging decomposition: 8 halves per thread (one K-group)
    const int sm = tid >> 2;                      // A row (0..63)
    const int sg = tid & 3;                       // K-group of 8 (0..3)
    const int pg = ((sg & 1) << 1) | (sg >> 1);   // permuted group: 0,2,1,3

    // W staging decomposition: k per lane, 8-n run per thread
    const int sk  = tid & 31;                     // k within tile (0..31)
    const int snr = (tid >> 5) * 8;               // n run start (0,8,..,56)

    v8f acc0 = {};
    v8f acc1 = {};

    for (int k0 = 0; k0 < lda; k0 += 32) {
        // ---- stage A: one global b128 + one LDS b128 per thread ----
        {
            int gm = mBase + sm;
            u32x4 va = {0u, 0u, 0u, 0u};
            if (gm < M) va = *(const u32x4*)(A + (size_t)gm * lda + k0 + sg * 8);
            *(u32x4*)(&As2[sm * 32 + pg * 8]) = va;
        }
        // ---- stage W: coalesced row reads, one b128 LDS store ----
        {
            Pk pk;
            if ((nBase + 64 <= N) && (k0 + 32 <= K)) {
                const float* wp = W + (size_t)(nBase + snr) * ldw + k0 + sk;
#pragma unroll
                for (int i = 0; i < 8; ++i) pk.s[i] = f2bf(wp[(size_t)i * ldw]);
                if (k0 + 32 < K) __builtin_prefetch(wp + 32, 0, 1);
            } else {
                int gk = k0 + sk;
#pragma unroll
                for (int i = 0; i < 8; ++i) {
                    int gn = nBase + snr + i;
                    float f = (gn < N && gk < K) ? W[(size_t)gn * ldw + gk] : 0.0f;
                    pk.s[i] = f2bf(f);
                }
            }
            *(u32x4*)(&Bs2[sk * 64 + snr]) = pk.q;
        }
        __syncthreads();

        // ---- fragments: A 2x ds_load_b128, B 16 u16 loads each ----
        Vb au, b0, b1;
        {
            const unsigned short* ap = &As2[(moff + (lane & 15)) * 32 + ((lane >> 4) << 4)];
            au.q[0] = *(const u32x4*)(ap);
            au.q[1] = *(const u32x4*)(ap + 8);
        }
        {
            const unsigned short* bp = &Bs2[noff + (lane & 15)];
            const int hh = lane >> 4;
#pragma unroll
            for (int s = 0; s < 16; ++s) {
                int kk = (s < 8) ? (8 * hh + s) : (16 + 8 * hh + (s - 8));
                b0.u[s] = bp[kk * 64];
                b1.u[s] = bp[kk * 64 + 16];
            }
        }

        acc0 = __builtin_amdgcn_wmma_f32_16x16x32_bf16(false, au.v, false, b0.v,
                                                       (short)0, acc0, false, false);
        acc1 = __builtin_amdgcn_wmma_f32_16x16x32_bf16(false, au.v, false, b1.v,
                                                       (short)0, acc1, false, false);
        __syncthreads();
    }

    // C layout: VGPR r, lanes 0-15 -> M=r, lanes 16-31 -> M=r+8; N = lane&15
#pragma unroll
    for (int r = 0; r < 8; ++r) {
        int m = mBase + moff + r + ((lane >> 4) << 3);
        if (m < M) {
            int n0 = nBase + noff + (lane & 15);
            if (n0 < N) {
                float v = acc0[r] + (bias ? bias[n0] : 0.0f);
                if (act) v = fmaxf(v, 0.0f);
                Cout[(size_t)m * N + n0] = v;
            }
            int n1 = n0 + 16;
            if (n1 < N) {
                float v = acc1[r] + (bias ? bias[n1] : 0.0f);
                if (act) v = fmaxf(v, 0.0f);
                Cout[(size_t)m * N + n1] = v;
            }
        }
    }
}

// ---------------- LSTM pointwise cell ----------------
__global__ __launch_bounds__(256) void lstm_cell_kernel(
    const float* __restrict__ XG, const float* __restrict__ HG,
    const float* __restrict__ bias, const float* __restrict__ mask,
    int t, int reverse,
    float* __restrict__ h, float* __restrict__ c,
    float* __restrict__ hcat, unsigned short* __restrict__ hcat_bf,
    unsigned short* __restrict__ hbf, int dirOff) {
    int idx = blockIdx.x * blockDim.x + threadIdx.x;
    if (idx >= B_ * H_) return;
    int b = idx / H_, j = idx % H_;
    int ta = reverse ? (T_ - 1 - t) : t;
    size_t row = (size_t)b * T_ + ta;
    const float* xg = XG + row * G4H_;
    const float* hg = HG + (size_t)b * G4H_;

    float gi = sigf(xg[j]            + hg[j]            + bias[j]);
    float gf = sigf(xg[H_ + j]       + hg[H_ + j]       + bias[H_ + j]);
    float gg = tanhf(xg[2 * H_ + j]  + hg[2 * H_ + j]   + bias[2 * H_ + j]);
    float go = sigf(xg[3 * H_ + j]   + hg[3 * H_ + j]   + bias[3 * H_ + j]);

    float cprev = c[idx];
    float c2 = gf * cprev + gi * gg;
    float h2 = go * tanhf(c2);
    float m = mask[row];
    float hn = m * h2 + (1.0f - m) * h[idx];
    float cn = m * c2 + (1.0f - m) * cprev;
    h[idx] = hn;
    c[idx] = cn;
    float y = h2 * m;
    hcat[row * D2_ + dirOff + j]     = y;
    hcat_bf[row * D2P_ + dirOff + j] = f2bf(y);
    hbf[(size_t)b * EP_ + j]         = f2bf(hn);   // H padded to 320 for next-step GEMM
}

// ---------------- final hidden concat ----------------
__global__ __launch_bounds__(256) void finalize_ht_kernel(
    const float* __restrict__ hf, const float* __restrict__ hb,
    float* __restrict__ ht, unsigned short* __restrict__ ht_bf) {
    int idx = blockIdx.x * blockDim.x + threadIdx.x;
    if (idx >= B_ * H_) return;
    int b = idx / H_, j = idx % H_;
    ht[(size_t)b * D2_ + j]          = hf[idx];
    ht[(size_t)b * D2_ + H_ + j]     = hb[idx];
    ht_bf[(size_t)b * D2P_ + j]      = f2bf(hf[idx]);
    ht_bf[(size_t)b * D2P_ + H_ + j] = f2bf(hb[idx]);
}

// ---------------- attention ----------------
__global__ __launch_bounds__(256) void alpha_kernel(const float* __restrict__ hcat,
                                                    const float* __restrict__ ht,
                                                    float* __restrict__ alpha) {
    int idx = blockIdx.x * blockDim.x + threadIdx.x;
    if (idx >= B_ * T_) return;
    int b = idx / T_;
    const float* hr = hcat + (size_t)idx * D2_;
    const float* hv = ht + (size_t)b * D2_;
    float s = 0.0f;
    for (int d = 0; d < D2_; ++d) s += hr[d] * hv[d];
    alpha[idx] = s;
}

__global__ __launch_bounds__(256) void h1_kernel(const float* __restrict__ hcat,
                                                 const float* __restrict__ alpha,
                                                 float* __restrict__ h1,
                                                 unsigned short* __restrict__ h1_bf) {
    int idx = blockIdx.x * blockDim.x + threadIdx.x;
    if (idx >= B_ * D2_) return;
    int b = idx / D2_, d = idx % D2_;
    float s = 0.0f;
    for (int t = 0; t < T_; ++t)
        s += hcat[((size_t)b * T_ + t) * D2_ + d] * alpha[b * T_ + t];
    h1[idx] = s;
    h1_bf[(size_t)b * D2P_ + d] = f2bf(s);
}

// ---------------- fp32 -> bf16 (row padded) ----------------
__global__ __launch_bounds__(256) void f32_to_bf16_pad_kernel(
    const float* __restrict__ src, unsigned short* __restrict__ dst,
    int rows, int K, int ldp) {
    int idx = blockIdx.x * blockDim.x + threadIdx.x;
    if (idx >= rows * ldp) return;
    int m = idx / ldp, k = idx % ldp;
    dst[idx] = (k < K) ? f2bf(src[(size_t)m * K + k]) : (unsigned short)0;
}

// ---------------- masked max over time ----------------
__global__ __launch_bounds__(256) void maxpos_kernel(const float* __restrict__ P,
                                                     int N, const float* __restrict__ mask,
                                                     unsigned short* __restrict__ dst_bf,
                                                     int ldp) {
    int idx = blockIdx.x * blockDim.x + threadIdx.x;
    if (idx >= B_ * N) return;
    int b = idx / N, n = idx % N;
    float best = -3.0e38f;
    for (int t = 0; t < T_; ++t) {
        float m = mask[b * T_ + t];
        float v = (m > 0.5f) ? P[((size_t)b * T_ + t) * N + n] : NEG_POS_;
        best = fmaxf(best, v);
    }
    dst_bf[(size_t)b * ldp + n] = f2bf(best);
}

// ---------------- per-row mean over C ----------------
__global__ __launch_bounds__(256) void rowmean_kernel(const float* __restrict__ P,
                                                      float* __restrict__ mean) {
    __shared__ float red[256];
    int b = blockIdx.x, t = threadIdx.x;
    float s = 0.0f;
    for (int n = t; n < C_; n += 256) s += P[(size_t)b * C_ + n];
    red[t] = s;
    __syncthreads();
    for (int off = 128; off > 0; off >>= 1) {
        if (t < off) red[t] += red[t + off];
        __syncthreads();
    }
    if (t == 0) mean[b] = red[0] / (float)C_;
}

// ---------------- score += P + mean[b]*msk[n] ----------------
__global__ __launch_bounds__(256) void combine_kernel(float* __restrict__ score,
                                                      const float* __restrict__ P,
                                                      const float* __restrict__ mean,
                                                      const float* __restrict__ msk) {
    size_t idx = (size_t)blockIdx.x * blockDim.x + threadIdx.x;
    if (idx >= (size_t)B_ * C_) return;
    int b = (int)(idx / C_);
    int n = (int)(idx % C_);
    score[idx] += P[idx] + mean[b] * msk[n];
}

// ---------------- scatter NEG_MSK at input-token columns ----------------
__global__ __launch_bounds__(256) void maskout_kernel(const int* __restrict__ x,
                                                      float* __restrict__ score) {
    int idx = blockIdx.x * blockDim.x + threadIdx.x;
    if (idx >= B_ * T_) return;
    int b = idx / T_;
    int xv = x[idx];
    int col = (xv < C_) ? xv : 0;
    score[(size_t)b * C_ + col] = NEG_MSK_;
}

// ---------------- radix argsort (descending, stable) ----------------
__global__ __launch_bounds__(256) void radix_init_kernel(const float* __restrict__ score,
                                                         unsigned* __restrict__ keys,
                                                         int* __restrict__ idx) {
    size_t i = (size_t)blockIdx.x * blockDim.x + threadIdx.x;
    if (i >= (size_t)B_ * C_) return;
    unsigned u = __float_as_uint(score[i]);
    unsigned k = (u & 0x80000000u) ? ~u : (u | 0x80000000u); // ascending-float key
    keys[i] = ~k;                                            // flip -> descending
    idx[i]  = (int)(i % C_);
}

#define RS_CHUNK ((C_ + 255) / 256)
__global__ __launch_bounds__(256) void radix_pass_kernel(const unsigned* __restrict__ kin,
                                                         const int* __restrict__ iin,
                                                         unsigned* __restrict__ kout,
                                                         int* __restrict__ iout,
                                                         int shift) {
    __shared__ unsigned cnt[256 * 16];
    __shared__ unsigned tot[16];
    __shared__ unsigned base[16];
    int b = blockIdx.x, t = threadIdx.x;
    const unsigned* krow = kin + (size_t)b * C_;
    const int* irow = iin + (size_t)b * C_;
    unsigned* korow = kout + (size_t)b * C_;
    int* iorow = iout + (size_t)b * C_;

    int begin = t * RS_CHUNK;
    int end = begin + RS_CHUNK; if (end > C_) end = C_;
#pragma unroll
    for (int d = 0; d < 16; ++d) cnt[t * 16 + d] = 0;
    for (int i = begin; i < end; ++i) {
        int d = (krow[i] >> shift) & 15;
        cnt[t * 16 + d]++;
    }
    __syncthreads();
    if (t < 16) {
        unsigned s = 0;
        for (int tt = 0; tt < 256; ++tt) s += cnt[tt * 16 + t];
        tot[t] = s;
    }
    __syncthreads();
    if (t == 0) {
        unsigned run = 0;
        for (int d = 0; d < 16; ++d) { base[d] = run; run += tot[d]; }
    }
    __syncthreads();
    if (t < 16) {
        unsigned run = base[t];
        for (int tt = 0; tt < 256; ++tt) {
            unsigned v = cnt[tt * 16 + t];
            cnt[tt * 16 + t] = run;
            run += v;
        }
    }
    __syncthreads();
    for (int i = begin; i < end; ++i) {
        unsigned k = krow[i];
        int d = (k >> shift) & 15;
        unsigned pos = cnt[t * 16 + d]++;
        korow[pos] = k;
        iorow[pos] = irow[i];
    }
}

__global__ __launch_bounds__(256) void write_idx_kernel(const int* __restrict__ idx,
                                                        int* __restrict__ out) {
    size_t i = (size_t)blockIdx.x * blockDim.x + threadIdx.x;
    if (i >= (size_t)B_ * C_) return;
    out[i] = idx[i];
}

// ======================= host launch =======================
extern "C" void kernel_launch(void* const* d_in, const int* in_sizes, int n_in,
                              void* d_out, int out_size, void* d_ws, size_t ws_size,
                              hipStream_t stream) {
    (void)in_sizes; (void)n_in; (void)out_size; (void)ws_size;

    const int*   x     = (const int*)  d_in[0];
    const float* ws_w  = (const float*)d_in[1];
    const float* wl_w  = (const float*)d_in[2];
    const float* wr_w  = (const float*)d_in[3];
    const float* msk_s = (const float*)d_in[4];
    const float* msk_l = (const float*)d_in[5];
    const float* msk_r = (const float*)d_in[6];
    const float* emb   = (const float*)d_in[7];
    const float* Wih_f = (const float*)d_in[8];
    const float* Whh_f = (const float*)d_in[9];
    const float* b_f   = (const float*)d_in[10];
    const float* Wih_b = (const float*)d_in[11];
    const float* Whh_b = (const float*)d_in[12];
    const float* b_b   = (const float*)d_in[13];
    const float* fc_W  = (const float*)d_in[14];
    const float* fc_b  = (const float*)d_in[15];
    const float* fcs_W = (const float*)d_in[16];
    const float* fcs_b = (const float*)d_in[17];
    const float* fcl_W = (const float*)d_in[18];
    const float* fcl_b = (const float*)d_in[19];
    const float* fcr_W = (const float*)d_in[20];
    const float* fcr_b = (const float*)d_in[21];

    float* score = (float*)d_out;                       // B x C
    int* outIdx = (int*)d_out + (size_t)B_ * C_;        // B x C indices

    // ---- workspace layout ----
    char* wsb = (char*)d_ws;
    size_t off = 0;
    auto alloc = [&](size_t bytes) -> void* {
        off = (off + 255) & ~(size_t)255;
        void* p = wsb + off;
        off += bytes;
        return p;
    };
    // big GEMM region (later aliased by radix ping-pong buffers)
    float* XGf = (float*)alloc((size_t)B_ * T_ * G4H_ * 4);
    float* XGb = (float*)alloc((size_t)B_ * T_ * G4H_ * 4);
    float* P   = (float*)alloc((size_t)B_ * T_ * S_ * 4);
    unsigned* keysA = (unsigned*)XGf;                    // aliases (sort runs last)
    int* idxA  = (int*)(keysA + (size_t)B_ * C_);
    unsigned* keysB = (unsigned*)(idxA + (size_t)B_ * C_);
    int* idxB  = (int*)(keysB + (size_t)B_ * C_);

    float* HGf = (float*)alloc((size_t)B_ * G4H_ * 4);
    float* HGb = (float*)alloc((size_t)B_ * G4H_ * 4);
    float* state = (float*)alloc((size_t)4 * B_ * H_ * 4);
    float* h_f = state;
    float* c_f = state + (size_t)B_ * H_;
    float* h_b = state + (size_t)2 * B_ * H_;
    float* c_b = state + (size_t)3 * B_ * H_;
    float* hcat  = (float*)alloc((size_t)B_ * T_ * D2_ * 4);
    float* ht    = (float*)alloc((size_t)B_ * D2_ * 4);
    float* alpha = (float*)alloc((size_t)B_ * T_ * 4);
    float* h1    = (float*)alloc((size_t)B_ * D2_ * 4);
    float* vd    = (float*)alloc((size_t)B_ * E_ * 4);
    float* t1    = (float*)alloc((size_t)B_ * L_ * 4);
    float* Ppath = (float*)alloc((size_t)B_ * C_ * 4);
    float* meanb = (float*)alloc((size_t)B_ * 4);
    float* maskf = (float*)alloc((size_t)B_ * T_ * 4);

    const size_t N_XEMB = (size_t)B_ * T_ * EP_;
    const size_t N_HBF  = (size_t)B_ * EP_;
    const size_t N_HCB  = (size_t)B_ * T_ * D2P_;
    const size_t N_HTB  = (size_t)B_ * D2P_;
    const size_t N_SEM  = (size_t)B_ * SP_;
    const size_t N_RA   = (size_t)B_ * RP_;
    const size_t N_T1   = (size_t)B_ * LP_;
    const size_t N_VD   = (size_t)B_ * EP_;
    const size_t BF_TOT = N_XEMB + 2 * N_HBF + N_HCB + 2 * N_HTB + N_VD + N_SEM + N_RA + N_T1;
    unsigned short* bfbase = (unsigned short*)alloc(BF_TOT * 2);
    unsigned short* xemb_bf = bfbase;
    unsigned short* hbf_f   = xemb_bf + N_XEMB;
    unsigned short* hbf_b   = hbf_f + N_HBF;
    unsigned short* hcat_bf = hbf_b + N_HBF;
    unsigned short* ht_bf   = hcat_bf + N_HCB;
    unsigned short* h1_bf   = ht_bf + N_HTB;
    unsigned short* vd_bf   = h1_bf + N_HTB;
    unsigned short* sem_bf  = vd_bf + N_VD;
    unsigned short* ra_bf   = sem_bf + N_SEM;
    unsigned short* t1_bf   = ra_bf + N_RA;

    auto gemm = [&](const unsigned short* A, int lda, const float* W, int ldw,
                    const float* bias, float* Cmat, int M, int N, int K, int act) {
        dim3 g((N + 63) / 64, (M + 63) / 64);
        gemm_bf16_kernel<<<g, 256, 0, stream>>>(A, lda, W, ldw, bias, Cmat, M, N, K, act);
    };

    // ---- init ----
    zero_u16_kernel<<<2048, 256, 0, stream>>>(bfbase, BF_TOT);
    zero_f32_kernel<<<512, 256, 0, stream>>>(state, (size_t)4 * B_ * H_);

    // ---- embedding + mask ----
    embed_kernel<<<B_ * T_, EP_, 0, stream>>>(x, emb, xemb_bf, maskf);

    // ---- input-gate GEMMs for both directions ----
    gemm(xemb_bf, EP_, Wih_f, E_, nullptr, XGf, B_ * T_, G4H_, E_, 0);
    gemm(xemb_bf, EP_, Wih_b, E_, nullptr, XGb, B_ * T_, G4H_, E_, 0);

    // ---- LSTM recurrence ----
    int nBH = (B_ * H_ + 255) / 256;
    for (int t = 0; t < T_; ++t) {
        gemm(hbf_f, EP_, Whh_f, H_, nullptr, HGf, B_, G4H_, H_, 0);
        lstm_cell_kernel<<<nBH, 256, 0, stream>>>(XGf, HGf, b_f, maskf, t, 0,
                                                  h_f, c_f, hcat, hcat_bf, hbf_f, 0);
        gemm(hbf_b, EP_, Whh_b, H_, nullptr, HGb, B_, G4H_, H_, 0);
        lstm_cell_kernel<<<nBH, 256, 0, stream>>>(XGb, HGb, b_b, maskf, t, 1,
                                                  h_b, c_b, hcat, hcat_bf, hbf_b, H_);
    }
    finalize_ht_kernel<<<nBH, 256, 0, stream>>>(h_f, h_b, ht, ht_bf);

    // ---- attention ----
    alpha_kernel<<<(B_ * T_ + 255) / 256, 256, 0, stream>>>(hcat, ht, alpha);
    h1_kernel<<<(B_ * D2_ + 255) / 256, 256, 0, stream>>>(hcat, alpha, h1, h1_bf);

    // ---- vd = h1 @ fc_W^T + fc_b ; score = vd @ emb[:C]^T ----
    gemm(h1_bf, D2P_, fc_W, D2_, fc_b, vd, B_, E_, D2_, 0);
    f32_to_bf16_pad_kernel<<<(B_ * EP_ + 255) / 256, 256, 0, stream>>>(vd, vd_bf, B_, E_, EP_);
    gemm(vd_bf, EP_, emb, E_, nullptr, score, B_, C_, E_, 0);

    size_t nBC = ((size_t)B_ * C_ + 255) / 256;

    // ---- s path ----
    gemm(hcat_bf, D2P_, fcs_W, D2_, fcs_b, P, B_ * T_, S_, D2_, 0);
    maxpos_kernel<<<(B_ * S_ + 255) / 256, 256, 0, stream>>>(P, S_, maskf, sem_bf, SP_);
    gemm(sem_bf, SP_, ws_w, S_, nullptr, Ppath, B_, C_, S_, 1);
    rowmean_kernel<<<B_, 256, 0, stream>>>(Ppath, meanb);
    combine_kernel<<<(unsigned)nBC, 256, 0, stream>>>(score, Ppath, meanb, msk_s);

    // ---- r path ----
    gemm(hcat_bf, D2P_, fcr_W, D2_, fcr_b, P, B_ * T_, R_, D2_, 0);
    maxpos_kernel<<<(B_ * R_ + 255) / 256, 256, 0, stream>>>(P, R_, maskf, ra_bf, RP_);
    gemm(ra_bf, RP_, wr_w, R_, nullptr, Ppath, B_, C_, R_, 1);
    rowmean_kernel<<<B_, 256, 0, stream>>>(Ppath, meanb);
    combine_kernel<<<(unsigned)nBC, 256, 0, stream>>>(score, Ppath, meanb, msk_r);

    // ---- l path ----
    gemm(h1_bf, D2P_, fcl_W, D2_, fcl_b, t1, B_, L_, D2_, 0);
    f32_to_bf16_pad_kernel<<<(B_ * LP_ + 255) / 256, 256, 0, stream>>>(t1, t1_bf, B_, L_, LP_);
    gemm(t1_bf, LP_, wl_w, L_, nullptr, Ppath, B_, C_, L_, 1);
    rowmean_kernel<<<B_, 256, 0, stream>>>(Ppath, meanb);
    combine_kernel<<<(unsigned)nBC, 256, 0, stream>>>(score, Ppath, meanb, msk_l);

    // ---- mask out input tokens ----
    maskout_kernel<<<(B_ * T_ + 255) / 256, 256, 0, stream>>>(x, score);

    // ---- descending stable argsort (8 x 4-bit LSD radix) ----
    radix_init_kernel<<<(unsigned)nBC, 256, 0, stream>>>(score, keysA, idxA);
    for (int p = 0; p < 8; ++p) {
        if ((p & 1) == 0)
            radix_pass_kernel<<<B_, 256, 0, stream>>>(keysA, idxA, keysB, idxB, p * 4);
        else
            radix_pass_kernel<<<B_, 256, 0, stream>>>(keysB, idxB, keysA, idxA, p * 4);
    }
    write_idx_kernel<<<(unsigned)nBC, 256, 0, stream>>>(idxA, outIdx);
}
